// BlockNet_14774687498746
// MI455X (gfx1250) — matrix operand order repaired and checked by reading, not compiled, not run on hardware
//
#include <hip/hip_runtime.h>
#include <hip/hip_bf16.h>

typedef __attribute__((ext_vector_type(16))) _Float16 v16h;
typedef __attribute__((ext_vector_type(8)))  float    v8f;
typedef __attribute__((ext_vector_type(4)))  unsigned int v4u;
typedef __attribute__((ext_vector_type(8)))  int      v8i;
typedef __attribute__((ext_vector_type(4)))  int      v4i;

#define NN 20000
#define NFEAT 128
#define NEDGE 320000
#define NQ 200000

// ---------------------------------------------------------------------------
// WMMA GEMM: C[M,N] = A[M,K] @ op(B).  f32 in / f16 WMMA / f32 acc.
// One wave computes a 32x32 C macro-tile: 2 A-frags x 2 B-frags -> 4
// independent v_wmma_f32_16x16x32_f16 chains per k-step (hides XDL RAW
// latency, 2x fragment reuse, 8 FLOP/byte from global).
// Requires M%32==0 and K%32==0 (true at every call site). N ragged: B column
// loads are pointer-clamped (garbage only reaches never-stored columns since
// D column n depends only on B column n), stores are guarded.
// ---------------------------------------------------------------------------
__global__ void gemm_wmma_k(const float* __restrict__ A, const float* __restrict__ B,
                            float* __restrict__ C,
                            int M, int N, int K, int lda, int ldb, int ldc, int transB)
{
    const int lane   = threadIdx.x & 31;
    const int wv     = threadIdx.x >> 5;
    const int tilesN = (N + 31) >> 5;
    const int tilesM = M >> 5;                       // M % 32 == 0
    const int tile   = blockIdx.x * (blockDim.x >> 5) + wv;
    if (tile >= tilesM * tilesN) return;             // wave-uniform: EXEC stays full
    const int tm = tile / tilesN, tn = tile % tilesN;
    const int m0 = tm << 5,       n0 = tn << 5;
    const int hi = lane >> 4;
    const int lr = lane & 15;
    const int nc0 = n0 + lr;                         // B/C columns, frag 0
    const int nc1 = n0 + 16 + lr;                    // frag 1
    const int bc0 = (nc0 < N) ? nc0 : 0;             // clamp: stay in-bounds
    const int bc1 = (nc1 < N) ? nc1 : 0;

    v8f acc00 = {}, acc01 = {}, acc10 = {}, acc11 = {};
    for (int k0 = 0; k0 < K; k0 += 32) {
        v16h a0 = {}, a1 = {}, b0 = {}, b1 = {};
        const float* Ap0 = A + (size_t)(m0 + lr) * lda + k0 + hi * 16;
        const float* Ap1 = Ap0 + (size_t)16 * lda;
        __builtin_prefetch(Ap0 + 32, 0, 1);          // global_prefetch_b8 next chunk
        __builtin_prefetch(Ap1 + 32, 0, 1);
#pragma unroll
        for (int e = 0; e < 8; ++e) {
            a0[e]     = (_Float16)Ap0[e];
            a0[e + 8] = (_Float16)Ap0[8 + e];
            a1[e]     = (_Float16)Ap1[e];
            a1[e + 8] = (_Float16)Ap1[8 + e];
        }
        if (!transB) {
            const float* Bp = B + (size_t)(k0 + hi * 16) * ldb;
#pragma unroll
            for (int e = 0; e < 16; ++e) {
                b0[e] = (_Float16)Bp[(size_t)e * ldb + bc0];
                b1[e] = (_Float16)Bp[(size_t)e * ldb + bc1];
            }
        } else {   // B stored [N,K] row-major; contiguous per lane
            const float* Bq0 = B + (size_t)bc0 * ldb + k0 + hi * 16;
            const float* Bq1 = B + (size_t)bc1 * ldb + k0 + hi * 16;
#pragma unroll
            for (int e = 0; e < 16; ++e) {
                b0[e] = (_Float16)Bq0[e];
                b1[e] = (_Float16)Bq1[e];
            }
        }
        acc00 = __builtin_amdgcn_wmma_f32_16x16x32_f16(false, a0, false, b0, (short)0, acc00, false, false);
        acc01 = __builtin_amdgcn_wmma_f32_16x16x32_f16(false, a0, false, b1, (short)0, acc01, false, false);
        acc10 = __builtin_amdgcn_wmma_f32_16x16x32_f16(false, a1, false, b0, (short)0, acc10, false, false);
        acc11 = __builtin_amdgcn_wmma_f32_16x16x32_f16(false, a1, false, b1, (short)0, acc11, false, false);
    }
    if (nc0 < N) {
#pragma unroll
        for (int r = 0; r < 8; ++r) {
            C[(size_t)(m0 + hi * 8 + r) * ldc + nc0]      = acc00[r];
            C[(size_t)(m0 + 16 + hi * 8 + r) * ldc + nc0] = acc10[r];
        }
    }
    if (nc1 < N) {
#pragma unroll
        for (int r = 0; r < 8; ++r) {
            C[(size_t)(m0 + hi * 8 + r) * ldc + nc1]      = acc01[r];
            C[(size_t)(m0 + 16 + hi * 8 + r) * ldc + nc1] = acc11[r];
        }
    }
}

// ---------------------------------------------------------------------------
// small utility kernels
// ---------------------------------------------------------------------------
__global__ void fill_k(float* p, int n, float v) {
    int i = blockIdx.x * blockDim.x + threadIdx.x;
    if (i < n) p[i] = v;
}
__global__ void deg_acc_k(const int* __restrict__ dst, float* deg, int nE) {
    int e = blockIdx.x * blockDim.x + threadIdx.x;
    if (e < nE) atomicAdd(&deg[dst[e]], 1.0f);
}
__global__ void rsqrt_k(float* p, int n) {
    int i = blockIdx.x * blockDim.x + threadIdx.x;
    if (i < n) p[i] = rsqrtf(p[i]);                  // deg >= 1 always (self loops)
}
__global__ void gcn_agg_k(const float* __restrict__ H, const int* __restrict__ src,
                          const int* __restrict__ dst, const float* __restrict__ dinv,
                          float* __restrict__ out, int nE, int nN, int F)
{
    int e = blockIdx.x * blockDim.x + threadIdx.x;
    if (e >= nE + nN) return;
    int s, d;
    if (e < nE) { s = src[e]; d = dst[e]; } else { s = d = e - nE; }
    float w = dinv[s] * dinv[d];
    const float* hs = H + (size_t)s * F;
    float* o = out + (size_t)d * F;
    for (int f = 0; f < F; ++f) atomicAdd(&o[f], hs[f] * w);
}
__global__ void bias_relu_k(float* A, const float* __restrict__ b, int n, int F) {
    int i = blockIdx.x * blockDim.x + threadIdx.x;
    if (i < n * F) {
        float v = A[i] + b[i % F];
        A[i] = v > 0.0f ? v : 0.0f;
    }
}
__global__ void renorm_k(float* V, int n, int F) {
    int i = blockIdx.x * blockDim.x + threadIdx.x;
    if (i >= n) return;
    float* r = V + (size_t)i * F;
    float s = 0.0f;
    for (int f = 0; f < F; ++f) s += r[f] * r[f];
    float nrm = sqrtf(s);
    float sc = (nrm > 1.0f) ? 1.0f / (nrm + 1e-7f) : 1.0f;
    for (int f = 0; f < F; ++f) r[f] *= sc;
}

// ---------------------------------------------------------------------------
// Build 1024x1024 block matrix [ L0r  rel_ ; rel_^T  L1r ], relu, row-softmax.
// ---------------------------------------------------------------------------
__global__ void softmax_rows_k(const float* __restrict__ L0r, const float* __restrict__ L1r,
                               const float* __restrict__ relx, float* __restrict__ S)
{
    __shared__ float red[256];
    const int row = blockIdx.x;
    const int t   = threadIdx.x;
    float v[4];
    float mx = 0.0f;                                 // relu output >= 0
#pragma unroll
    for (int i = 0; i < 4; ++i) {
        int j = t + i * 256;
        float raw;
        if (row < 512) raw = (j < 512) ? L0r[(size_t)row * 512 + j]
                                       : relx[(size_t)row * 512 + (j - 512)];
        else {
            int r = row - 512;
            raw = (j < 512) ? relx[(size_t)j * 512 + r]          // rel_^T
                            : L1r[(size_t)r * 512 + (j - 512)];
        }
        v[i] = raw > 0.0f ? raw : 0.0f;
        mx = fmaxf(mx, v[i]);
    }
    red[t] = mx; __syncthreads();
    for (int s2 = 128; s2 > 0; s2 >>= 1) {
        if (t < s2) red[t] = fmaxf(red[t], red[t + s2]);
        __syncthreads();
    }
    mx = red[0]; __syncthreads();
    float sum = 0.0f;
#pragma unroll
    for (int i = 0; i < 4; ++i) { v[i] = expf(v[i] - mx); sum += v[i]; }
    red[t] = sum; __syncthreads();
    for (int s2 = 128; s2 > 0; s2 >>= 1) {
        if (t < s2) red[t] += red[t + s2];
        __syncthreads();
    }
    float inv = 1.0f / red[0];
#pragma unroll
    for (int i = 0; i < 4; ++i) S[(size_t)row * 1024 + t + i * 256] = v[i] * inv;
}

// ---------------------------------------------------------------------------
// TDM: DMA a 1-D run of f32 (nelem elements) from global into LDS.
// D# built per CDNA5 ISA §8: group0 {count=1, lds_addr, global_addr, type=2},
// group1 {data_size=4B, tensor_dim0=nelem, tensor_dim1=1, tile_dim0=nelem,
// tile_dim1=1, dim0_stride=nelem}. Trailing groups zero (<=2D tensor).
// Tracked by TENSORcnt; issuing wave waits s_wait_tensorcnt 0.
// ---------------------------------------------------------------------------
__device__ __forceinline__ void tdm_load_1d_f32(unsigned int lds_addr,
                                                const float* gsrc,
                                                unsigned int nelem)
{
    unsigned long long ga = (unsigned long long)(uintptr_t)gsrc;
    v4u g0 = {};
    g0[0] = 1u;                                          // count=1 (valid user D#)
    g0[1] = lds_addr;                                    // bits [63:32]  lds byte addr
    g0[2] = (unsigned int)(ga & 0xFFFFFFFFu);            // bits [95:64]  gaddr lo
    g0[3] = (unsigned int)((ga >> 32) & 0x1FFFFFFu)      // bits [120:96] gaddr hi
          | (2u << 30);                                  // bits [127:126] type=2
    v8i g1 = {};
    g1[0] = (int)(2u << 16);                             // data_size=2 -> 4 bytes
    g1[1] = (int)((nelem & 0xFFFFu) << 16);              // tensor_dim0 lo16 @ [63:48]
    g1[2] = (int)((nelem >> 16) & 0xFFFFu)               // tensor_dim0 hi16 @ [79:64]
          | (1 << 16);                                   // tensor_dim1=1   @ [95:80]
    g1[3] = (int)((nelem & 0xFFFFu) << 16);              // tile_dim0       @ [127:112]
    g1[4] = 1;                                           // tile_dim1=1     @ [143:128]
    g1[5] = (int)nelem;                                  // dim0_stride lo32
    g1[6] = 0;
    g1[7] = 0;
    v4i z4 = {};
    v8i z8 = {};
    __builtin_amdgcn_tensor_load_to_lds(g0, g1, z4, z4, z8, 0);   // 6-arg form
}

// ---------------------------------------------------------------------------
// Query head: thread per query. lin1_W staged to LDS via TDM (one async DMA
// per workgroup instead of 200K threads re-reading the same 5KB panel);
// LDS reads are same-address broadcasts (conflict-free).
// ---------------------------------------------------------------------------
__global__ void query_k(const float* __restrict__ g, const float* __restrict__ s,
                        const int* __restrict__ te,
                        const float* __restrict__ W1, const float* __restrict__ b1,
                        const float* __restrict__ W2, const float* __restrict__ b2,
                        float* __restrict__ out, int nq)
{
    __shared__ float W1s[40 * 32];
    if ((threadIdx.x >> 5) == 0) {                   // wave 0 issues the TDM DMA
        tdm_load_1d_f32((unsigned int)(uintptr_t)W1s, W1, 40 * 32);
        __builtin_amdgcn_s_wait_tensorcnt(0);
    }
    __syncthreads();

    int q = blockIdx.x * blockDim.x + threadIdx.x;
    if (q >= nq) return;
    int i0 = te[2 * q], i1 = te[2 * q + 1];
    float feat[40];
    const float* g0 = g + (size_t)i0 * 32;
    const float* g1 = g + (size_t)i1 * 32;
#pragma unroll
    for (int k = 0; k < 32; ++k) { float d = g0[k] - g1[k]; feat[k] = 1.0f * d * d; }
    const float* s0 = s + (size_t)i0 * 8;
    const float* s1 = s + (size_t)i1 * 8;
#pragma unroll
    for (int k = 0; k < 8; ++k) { float d = s0[k] - s1[k]; feat[32 + k] = 0.1f * d * d; }
    float acc2 = b2[0];
    for (int j = 0; j < 32; ++j) {
        float a = b1[j];
#pragma unroll
        for (int k = 0; k < 40; ++k) a = fmaf(feat[k], W1s[k * 32 + j], a);
        a = (a > 0.0f) ? a : 0.2f * a;               // leaky_relu
        acc2 = fmaf(a, W2[j], acc2);
    }
    float sq = fminf(fabsf(acc2), 40.0f);
    out[q] = 1.0f / (1.0f + expf(sq - 2.0f));        // sigmoid(2 - sq)
}

// ---------------------------------------------------------------------------
extern "C" void kernel_launch(void* const* d_in, const int* in_sizes, int n_in,
                              void* d_out, int out_size, void* d_ws, size_t ws_size,
                              hipStream_t stream)
{
    const float* x       = (const float*)d_in[0];
    const float* L0      = (const float*)d_in[1];
    const float* L1      = (const float*)d_in[2];
    const int*   eidx    = (const int*)  d_in[3];   // [2, NEDGE]: src row, dst row
    const int*   tedges  = (const int*)  d_in[4];   // [NQ, 2]
    const float* W1      = (const float*)d_in[5];
    const float* b1      = (const float*)d_in[6];
    const float* W2      = (const float*)d_in[7];
    const float* b2      = (const float*)d_in[8];
    const float* w_sim   = (const float*)d_in[9];   // [1024, 8]
    const float* emb_sim = (const float*)d_in[10];  // [128, 1024]
    const float* w_off   = (const float*)d_in[11];  // [512, 512]
    const float* wL0     = (const float*)d_in[12];  // [512, 32]
    const float* wL1     = (const float*)d_in[13];  // [512, 32]
    const float* lin1W   = (const float*)d_in[14];  // [40, 32]
    const float* lin1b   = (const float*)d_in[15];
    const float* linW    = (const float*)d_in[16];  // [32, 1]
    const float* linb    = (const float*)d_in[17];
    float* out = (float*)d_out;
    (void)in_sizes; (void)n_in; (void)out_size; (void)ws_size;

    char* ws = (char*)d_ws;
    size_t off = 0;
    auto take = [&](size_t bytes) {
        char* p = ws + off;
        off += (bytes + 255) & ~(size_t)255;
        return p;
    };
    float* dinv = (float*)take((size_t)NN * 4);            // deg -> dinv in place
    float* H0   = (float*)take((size_t)NN * 64 * 4);
    float* A1   = (float*)take((size_t)NN * 64 * 4);       // agg1 -> h1
    float* H1   = (float*)take((size_t)NN * 32 * 4);
    float* A2   = (float*)take((size_t)NN * 32 * 4);       // agg2 -> g_emb -> g
    float* L0r  = (float*)take((size_t)512 * 512 * 4);
    float* L1r  = (float*)take((size_t)512 * 512 * 4);
    float* P0   = (float*)take((size_t)512 * 32 * 4);
    float* P1   = (float*)take((size_t)512 * 32 * 4);
    float* rel  = (float*)take((size_t)512 * 512 * 4);
    float* rel2 = (float*)take((size_t)512 * 512 * 4);
    float* S    = (float*)take((size_t)1024 * 1024 * 4);
    float* T1   = (float*)take((size_t)128 * 1024 * 4);
    float* Mm   = (float*)take((size_t)128 * 8 * 4);
    float* semb = (float*)take((size_t)NN * 8 * 4);

    auto ggrid = [](int M, int N) {
        int tiles = (M / 32) * ((N + 31) / 32);            // 32x32 macro-tiles
        return (tiles + 7) / 8;                            // 8 waves / 256-thread block
    };
    const int T = 256;

    // ---- degree / normalization ----
    fill_k<<<(NN + T - 1) / T, T, 0, stream>>>(dinv, NN, 1.0f);          // self loop
    deg_acc_k<<<(NEDGE + T - 1) / T, T, 0, stream>>>(eidx + NEDGE, dinv, NEDGE);
    rsqrt_k<<<(NN + T - 1) / T, T, 0, stream>>>(dinv, NN);
    fill_k<<<((NN * 64) + T - 1) / T, T, 0, stream>>>(A1, NN * 64, 0.0f);
    fill_k<<<((NN * 32) + T - 1) / T, T, 0, stream>>>(A2, NN * 32, 0.0f);

    // ---- GCN layer 1: H0 = x @ W1 ; scatter ; +b, relu ----
    gemm_wmma_k<<<ggrid(NN, 64), T, 0, stream>>>(x, W1, H0, NN, 64, NFEAT, NFEAT, 64, 64, 0);
    gcn_agg_k<<<(NEDGE + NN + T - 1) / T, T, 0, stream>>>(H0, eidx, eidx + NEDGE, dinv, A1, NEDGE, NN, 64);
    bias_relu_k<<<((NN * 64) + T - 1) / T, T, 0, stream>>>(A1, b1, NN, 64);

    // ---- GCN layer 2 ----
    gemm_wmma_k<<<ggrid(NN, 32), T, 0, stream>>>(A1, W2, H1, NN, 32, 64, 64, 32, 32, 0);
    gcn_agg_k<<<(NEDGE + NN + T - 1) / T, T, 0, stream>>>(H1, eidx, eidx + NEDGE, dinv, A2, NEDGE, NN, 32);
    bias_relu_k<<<((NN * 32) + T - 1) / T, T, 0, stream>>>(A2, b2, NN, 32);

    // ---- similarity block ----
    gemm_wmma_k<<<ggrid(512, 512), T, 0, stream>>>(L0, L0, L0r, 512, 512, 512, 512, 512, 512, 0);
    gemm_wmma_k<<<ggrid(512, 512), T, 0, stream>>>(L1, L1, L1r, 512, 512, 512, 512, 512, 512, 0);
    gemm_wmma_k<<<ggrid(512, 32),  T, 0, stream>>>(L0r, wL0, P0, 512, 32, 512, 512, 32, 32, 0);
    gemm_wmma_k<<<ggrid(512, 32),  T, 0, stream>>>(L1r, wL1, P1, 512, 32, 512, 512, 32, 32, 0);
    gemm_wmma_k<<<ggrid(512, 512), T, 0, stream>>>(P0, P1, rel, 512, 512, 32, 32, 32, 512, 1);   // B^T
    gemm_wmma_k<<<ggrid(512, 512), T, 0, stream>>>(w_off, rel, rel2, 512, 512, 512, 512, 512, 512, 0);
    softmax_rows_k<<<1024, 256, 0, stream>>>(L0r, L1r, rel2, S);

    // ---- re-associated embedding path: Mm = (emb @ S) @ w_sim ; semb = x @ Mm
    gemm_wmma_k<<<ggrid(128, 1024), T, 0, stream>>>(emb_sim, S, T1, 128, 1024, 1024, 1024, 1024, 1024, 0);
    gemm_wmma_k<<<ggrid(128, 8),    T, 0, stream>>>(T1, w_sim, Mm, 128, 8, 1024, 1024, 8, 8, 0);
    gemm_wmma_k<<<ggrid(NN, 8),     T, 0, stream>>>(x, Mm, semb, NN, 8, NFEAT, NFEAT, 8, 8, 0);

    // ---- row renorms ----
    renorm_k<<<(NN + T - 1) / T, T, 0, stream>>>(semb, NN, 8);
    renorm_k<<<(NN + T - 1) / T, T, 0, stream>>>(A2, NN, 32);

    // ---- query head ----
    query_k<<<(NQ + T - 1) / T, T, 0, stream>>>(A2, semb, tedges, lin1W, lin1b, linW, linb, out, NQ);
}
